// QKVW8A8Linear_927712936113
// MI455X (gfx1250) — 8-bit matrix operand order repaired and checked by reading, compiled, not run on hardware
//
#include <hip/hip_runtime.h>
#include <stdint.h>

typedef __attribute__((ext_vector_type(8))) int v8i;

#define QEPS 1e-8f

__global__ void __launch_bounds__(256)
quant_rows_927712936113(const float* __restrict__ x,
                        int8_t* __restrict__ xq,
                        float* __restrict__ sx,
                        int K  ) {
    const int t   = blockIdx.x;
    const int tid = threadIdx.x;
    const float* xrow = x + (size_t)t * K;
    int8_t*      qrow = xq + (size_t)t * K;

    float4 v[4];
    float amax = 0.0f;
#pragma unroll
    for (int j = 0; j < 4; ++j) {
        const int base = tid * 4 + j * 1024;
        v[j] = *(const float4*)(xrow + base);
        amax = fmaxf(amax, fabsf(v[j].x));
        amax = fmaxf(amax, fabsf(v[j].y));
        amax = fmaxf(amax, fabsf(v[j].z));
        amax = fmaxf(amax, fabsf(v[j].w));
    }

#pragma unroll
    for (int off = 16; off > 0; off >>= 1)
        amax = fmaxf(amax, __shfl_xor(amax, off, 32));

    __shared__ float wmax[8];
    if ((tid & 31) == 0) wmax[tid >> 5] = amax;
    __syncthreads();

    float m = wmax[0];
#pragma unroll
    for (int i = 1; i < 8; ++i) m = fmaxf(m, wmax[i]);

    const float s   = fmaxf(m, QEPS) * (1.0f / 127.0f);
    const float inv = 1.0f / s;

#pragma unroll
    for (int j = 0; j < 4; ++j) {
        const int base = tid * 4 + j * 1024;
        float q0 = fminf(fmaxf(rintf(v[j].x * inv), -128.0f), 127.0f);
        float q1 = fminf(fmaxf(rintf(v[j].y * inv), -128.0f), 127.0f);
        float q2 = fminf(fmaxf(rintf(v[j].z * inv), -128.0f), 127.0f);
        float q3 = fminf(fmaxf(rintf(v[j].w * inv), -128.0f), 127.0f);
        int packed = ((int)q0 & 0xff) | (((int)q1 & 0xff) << 8) |
                     (((int)q2 & 0xff) << 16) | (((int)q3 & 0xff) << 24);
        *(int*)(qrow + base) = packed;
    }
    if (tid == 0) sx[t] = s;
}

__global__ void __launch_bounds__(256)
w8a8_gemm_927712936113(const int8_t* __restrict__ xq,
                       const int8_t* __restrict__ wgt,
                       const float*  __restrict__ sx,
                       const float*  __restrict__ scale,
                       const float*  __restrict__ bias,
                       float*        __restrict__ out,
                       int T, int N, int K) {
    const int lane = threadIdx.x & 31;
    const int wave = threadIdx.x >> 5;
    const int wm   = wave & 1;
    const int wn   = wave >> 1;

    const int mbase = blockIdx.y * 64 + wm * 32;
    const int nbase = blockIdx.x * 256 + wn * 64;

    const int l15 = lane & 15;
    const int lhi = lane >> 4;

    const int8_t* aptr0 = xq + (size_t)(mbase + l15) * K + lhi * 16;
    const int8_t* aptr1 = aptr0 + (size_t)16 * K;

    const int8_t* bptr0 = wgt + (size_t)(nbase + 0 * 16 + l15) * K + lhi * 16;
    const int8_t* bptr1 = wgt + (size_t)(nbase + 1 * 16 + l15) * K + lhi * 16;
    const int8_t* bptr2 = wgt + (size_t)(nbase + 2 * 16 + l15) * K + lhi * 16;
    const int8_t* bptr3 = wgt + (size_t)(nbase + 3 * 16 + l15) * K + lhi * 16;
    const int8_t* bptr[4] = {bptr0, bptr1, bptr2, bptr3};

    v8i acc[2][4];
#pragma unroll
    for (int i = 0; i < 2; ++i)
#pragma unroll
        for (int t = 0; t < 4; ++t) acc[i][t] = (v8i)0;

    for (int k0 = 0; k0 < K; k0 += 64) {
        v8i a0, a1;
        {
            const int8_t* p = aptr0 + k0;
            int4 t0 = *(const int4*)(p + 0);
            int4 t1 = *(const int4*)(p + 32);
            a0 = (v8i){t0.x, t0.y, t0.z, t0.w, t1.x, t1.y, t1.z, t1.w};
        }
        {
            const int8_t* p = aptr1 + k0;
            int4 t0 = *(const int4*)(p + 0);
            int4 t1 = *(const int4*)(p + 32);
            a1 = (v8i){t0.x, t0.y, t0.z, t0.w, t1.x, t1.y, t1.z, t1.w};
        }
#pragma unroll
        for (int t = 0; t < 4; ++t) {
            const int8_t* p = bptr[t] + k0;
            int4 u0 = *(const int4*)(p + 0);
            int4 u1 = *(const int4*)(p + 32);
            v8i b = (v8i){u0.x, u0.y, u0.z, u0.w, u1.x, u1.y, u1.z, u1.w};
            acc[0][t] = __builtin_amdgcn_wmma_i32_16x16x64_iu8(
                true, a0, true, b, acc[0][t], false, false);
            acc[1][t] = __builtin_amdgcn_wmma_i32_16x16x64_iu8(
                true, a1, true, b, acc[1][t], false, false);
        }
    }

#pragma unroll
    for (int i = 0; i < 2; ++i) {
        float srow[8];
#pragma unroll
        for (int r = 0; r < 8; ++r)
            srow[r] = sx[mbase + i * 16 + lhi * 8 + r];
#pragma unroll
        for (int t = 0; t < 4; ++t) {
            const int col = nbase + t * 16 + l15;
            const float ws = scale[col];
            const float bz = bias[col];
#pragma unroll
            for (int r = 0; r < 8; ++r) {
                const int row = mbase + i * 16 + lhi * 8 + r;
                out[(size_t)row * N + col] =
                    (float)acc[i][t][r] * srow[r] * ws + bz;
            }
        }
    }
}

extern "C" void kernel_launch(void* const* d_in, const int* in_sizes, int n_in,
                              void* d_out, int out_size, void* d_ws, size_t ws_size,
                              hipStream_t stream) {
    const float*  x     = (const float*)d_in[0];
    const int8_t* wgt   = (const int8_t*)d_in[1];
    const float*  scale = (const float*)d_in[2];
    const float*  bias  = (const float*)d_in[3];
    float*        out   = (float*)d_out;

    const int d_out_f = in_sizes[2];
    const int d_in_f  = in_sizes[1] / d_out_f;
    const int tokens  = in_sizes[0] / d_in_f;

    int8_t* xq = (int8_t*)d_ws;
    float*  sx = (float*)((char*)d_ws + (size_t)tokens * d_in_f);

    quant_rows_927712936113<<<tokens, 256, 0, stream>>>(x, xq, sx, d_in_f);

    dim3 grid(d_out_f / 256, tokens / 64);
    w8a8_gemm_927712936113<<<grid, 256, 0, stream>>>(
        xq, wgt, sx, scale, bias, out, tokens, d_out_f, d_in_f);
}
